// Block_4509715660807
// MI455X (gfx1250) — matrix operand order repaired and checked by reading, not compiled, not run on hardware
//
#include <hip/hip_runtime.h>
#include <math.h>

#define Bn 2
#define Sn 2048
#define Dn 1024
#define Hn 16
#define DHn 64
#define FFn 4096
#define TOKn (Bn * Sn)            // 4096 tokens
#define BSDn (Bn * Sn * Dn)

typedef __attribute__((ext_vector_type(16))) __bf16 v16bf;
typedef __attribute__((ext_vector_type(8)))  float  v8f;

static __device__ __forceinline__ __bf16 f2bf(float f) { return (__bf16)f; }

static __device__ __forceinline__ v8f wmma_bf16(v16bf a, v16bf b, v8f c) {
  return __builtin_amdgcn_wmma_f32_16x16x32_bf16(false, a, false, b, (short)0, c,
                                                 false, false);
}

static __device__ __forceinline__ float gelu_exact(float x) {
  return 0.5f * x * (1.0f + erff(x * 0.70710678118654752f));
}

#define ZERO8 ((v8f){0.f, 0.f, 0.f, 0.f, 0.f, 0.f, 0.f, 0.f})

enum { OUT_F32 = 0, OUT_BF16 = 1, OUT_BF16_T = 2 };

// ---------------------------------------------------------------------------
// Elementwise f32 -> bf16 (4 elements/thread)
// ---------------------------------------------------------------------------
__global__ __launch_bounds__(256)
void cvt_bf16_kernel(const float* __restrict__ in, __bf16* __restrict__ out) {
  const size_t i4 = (size_t)blockIdx.x * 256 + threadIdx.x;
  const float4 v = ((const float4*)in)[i4];
  out[i4 * 4 + 0] = f2bf(v.x);
  out[i4 * 4 + 1] = f2bf(v.y);
  out[i4 * 4 + 2] = f2bf(v.z);
  out[i4 * 4 + 3] = f2bf(v.w);
}

// ---------------------------------------------------------------------------
// W[K][N] f32  ->  Wt[N][K] bf16  (32x32 LDS-tiled transpose)
// ---------------------------------------------------------------------------
__global__ __launch_bounds__(256)
void transpose_cvt_kernel(const float* __restrict__ W, __bf16* __restrict__ Wt,
                          int K, int N) {
  __shared__ float tile[32][33];
  const int n0 = blockIdx.x * 32;
  const int k0 = blockIdx.y * 32;
  const int tx = threadIdx.x;          // 0..31
  const int ty = threadIdx.y;          // 0..7
#pragma unroll
  for (int i = 0; i < 4; ++i) {
    int k = k0 + ty + i * 8;
    tile[ty + i * 8][tx] = W[(size_t)k * N + n0 + tx];
  }
  __syncthreads();
#pragma unroll
  for (int i = 0; i < 4; ++i) {
    int n = n0 + ty + i * 8;
    Wt[(size_t)n * K + k0 + tx] = f2bf(tile[tx][ty + i * 8]);
  }
}

// ---------------------------------------------------------------------------
// Register-blocked bf16 GEMM: C[M,N] = act(A[M,K] @ W[K,N] + bias[N])
// A: bf16 row-major [M][K].  Wt: bf16 [N][K] (W transposed) so B-fragments
// are contiguous.  One wave computes 64x64: 16 v_wmma per K=32 step against
// 16 contiguous global_load_b128 (no conversions in the hot loop).
// OUTMODE: f32 [M][N] | bf16 [M][N] | bf16 transposed [N][M].
// ---------------------------------------------------------------------------
template <bool GELU, int OUTMODE>
__global__ __launch_bounds__(32)
void gemm_bias_kernel(const __bf16* __restrict__ A, const __bf16* __restrict__ Wt,
                      const float* __restrict__ bias, void* __restrict__ Cout,
                      int M, int N, int K) {
  constexpr int MT = 4, NT = 4;
  const int lane = threadIdx.x;
  const int half = lane >> 4;
  const int l16  = lane & 15;
  const int n0   = blockIdx.x * (16 * NT);
  const int row0 = blockIdx.y * (16 * MT);

  const int abase = half ? 16 : 0;   // A frag K offsets: {0..15} / {16..31}
  const int bbase = half ? 16 : 0;   // B frag K offsets: {0..15} / {16..31}

  v8f acc[MT][NT];
#pragma unroll
  for (int mt = 0; mt < MT; ++mt)
#pragma unroll
    for (int nt = 0; nt < NT; ++nt) acc[mt][nt] = ZERO8;

  for (int k0 = 0; k0 < K; k0 += 32) {
    v16bf a[MT], b[NT];
#pragma unroll
    for (int mt = 0; mt < MT; ++mt) {
      const __bf16* ak = A + (size_t)(row0 + mt * 16 + l16) * K + k0 + abase;
#pragma unroll
      for (int i = 0; i < 8; ++i) a[mt][i] = ak[i];
#pragma unroll
      for (int i = 0; i < 8; ++i) a[mt][8 + i] = ak[8 + i];
    }
#pragma unroll
    for (int nt = 0; nt < NT; ++nt) {
      const __bf16* bk = Wt + (size_t)(n0 + nt * 16 + l16) * K + k0 + bbase;
#pragma unroll
      for (int j = 0; j < 16; ++j) b[nt][j] = bk[j];
    }
#pragma unroll
    for (int mt = 0; mt < MT; ++mt)
#pragma unroll
      for (int nt = 0; nt < NT; ++nt)
        acc[mt][nt] = wmma_bf16(a[mt], b[nt], acc[mt][nt]);
  }

#pragma unroll
  for (int nt = 0; nt < NT; ++nt) {
    const int col = n0 + nt * 16 + l16;
    const float bv = bias[col];
#pragma unroll
    for (int mt = 0; mt < MT; ++mt)
#pragma unroll
      for (int e = 0; e < 8; ++e) {
        const int row = row0 + mt * 16 + e + (half ? 8 : 0);
        float r = acc[mt][nt][e] + bv;
        if (GELU) r = gelu_exact(r);
        if (OUTMODE == OUT_F32)
          ((float*)Cout)[(size_t)row * N + col] = r;
        else if (OUTMODE == OUT_BF16)
          ((__bf16*)Cout)[(size_t)row * N + col] = f2bf(r);
        else  // OUT_BF16_T: [N][M]; per-lane stores contiguous in e
          ((__bf16*)Cout)[(size_t)col * M + row] = f2bf(r);
      }
  }
}

// ---------------------------------------------------------------------------
// Flash attention: one wave per (batch, head, 16-query tile).
// q,k: bf16 [token][D]; vt: bf16 [D][token] (so P@V B-frags are contiguous).
// Per 32-key block: 4 WMMA (Q@K^T) + 1 WMMA (row sums = P@ones) + 4 WMMA (P@V).
// ctx written bf16 [token][D] for the O-projection.
// ---------------------------------------------------------------------------
__global__ __launch_bounds__(32)
void attn_kernel(const __bf16* __restrict__ q, const __bf16* __restrict__ kk,
                 const __bf16* __restrict__ vt, const int* __restrict__ mask,
                 __bf16* __restrict__ ctx) {
  const int lane = threadIdx.x;
  const int half = lane >> 4;
  const int l16  = lane & 15;
  const int q0 = blockIdx.x * 16;
  const int h  = blockIdx.y;
  const int b  = blockIdx.z;
  const int abase = half ? 16 : 0;
  const int bbase = half ? 16 : 0;
  const size_t hoff = (size_t)h * DHn;

  __shared__ __bf16 pbuf[16 * 32];

  v16bf ones;
#pragma unroll
  for (int j = 0; j < 16; ++j) ones[j] = f2bf(1.0f);

  // Q A-fragments: two K=32 chunks covering DH=64 (contiguous bf16 loads)
  v16bf aq[2];
  {
    const __bf16* qp = q + ((size_t)(b * Sn + q0 + l16)) * Dn + hoff;
#pragma unroll
    for (int c = 0; c < 2; ++c) {
      const __bf16* qc = qp + 32 * c + abase;
#pragma unroll
      for (int i = 0; i < 8; ++i) aq[c][i] = qc[i];
#pragma unroll
      for (int i = 0; i < 8; ++i) aq[c][8 + i] = qc[8 + i];
    }
  }

  float mrow[8], lrow[8], alpha[8];
#pragma unroll
  for (int e = 0; e < 8; ++e) { mrow[e] = -1e30f; lrow[e] = 0.f; }
  v8f co[4];
#pragma unroll
  for (int n = 0; n < 4; ++n) co[n] = ZERO8;

  for (int kt = 0; kt < Sn; kt += 32) {
    // scores: S(16q x 32k) = Q(16x64) @ K^T, two 16-key tiles, chained K-dim
    v8f s[2];
#pragma unroll
    for (int t = 0; t < 2; ++t) {
      s[t] = ZERO8;
      const int key = kt + 16 * t + l16;
      const __bf16* kp = kk + ((size_t)(b * Sn + key)) * Dn + hoff + bbase;
#pragma unroll
      for (int c = 0; c < 2; ++c) {
        v16bf bf;
        const __bf16* kc = kp + 32 * c;
#pragma unroll
        for (int j = 0; j < 16; ++j) bf[j] = kc[j];
        s[t] = wmma_bf16(aq[c], bf, s[t]);
      }
    }

    // scale + additive mask penalty
    float pen[2];
#pragma unroll
    for (int t = 0; t < 2; ++t) {
      float mv = (float)mask[b * Sn + kt + 16 * t + l16];
      pen[t] = 10000.f * (1.f - mv);
    }
#pragma unroll
    for (int t = 0; t < 2; ++t)
#pragma unroll
      for (int e = 0; e < 8; ++e)
        s[t][e] = s[t][e] * 0.125f - pen[t];   // 1/sqrt(64)

    // running max per row (cross-lane within each 16-lane half), then exp
#pragma unroll
    for (int e = 0; e < 8; ++e) {
      float mx = fmaxf(s[0][e], s[1][e]);
#pragma unroll
      for (int o = 8; o >= 1; o >>= 1) mx = fmaxf(mx, __shfl_xor(mx, o, 32));
      float mnew = fmaxf(mrow[e], mx);
      alpha[e]   = __expf(mrow[e] - mnew);
      mrow[e]    = mnew;
      s[0][e] = __expf(s[0][e] - mnew);
      s[1][e] = __expf(s[1][e] - mnew);
    }

    // re-stage P (C-frag layout) -> LDS (bf16) -> A-frag layout
#pragma unroll
    for (int e = 0; e < 8; ++e) {
      pbuf[(e + (half ? 8 : 0)) * 32 + l16]      = f2bf(s[0][e]);
      pbuf[(e + (half ? 8 : 0)) * 32 + 16 + l16] = f2bf(s[1][e]);
    }
    __syncthreads();
    v16bf pa;
    {
      const __bf16* pr = &pbuf[l16 * 32 + abase];
#pragma unroll
      for (int i = 0; i < 8; ++i) pa[i] = pr[i];
#pragma unroll
      for (int i = 0; i < 8; ++i) pa[8 + i] = pr[8 + i];
    }
    __syncthreads();

    // row sums of P via WMMA: every column of the C tile holds the row sum
    v8f sumt = wmma_bf16(pa, ones, ZERO8);
#pragma unroll
    for (int e = 0; e < 8; ++e) {
      lrow[e] = lrow[e] * alpha[e] + sumt[e];
#pragma unroll
      for (int n = 0; n < 4; ++n) co[n][e] *= alpha[e];
    }

    // out += P(16x32) @ V(32x64); V transposed => contiguous B-frags
#pragma unroll
    for (int n = 0; n < 4; ++n) {
      v16bf bv;
      const __bf16* vp =
          vt + (hoff + n * 16 + l16) * (size_t)TOKn + b * Sn + kt + bbase;
#pragma unroll
      for (int j = 0; j < 16; ++j) bv[j] = vp[j];
      co[n] = wmma_bf16(pa, bv, co[n]);
    }
  }

#pragma unroll
  for (int n = 0; n < 4; ++n)
#pragma unroll
    for (int e = 0; e < 8; ++e) {
      int row = q0 + e + (half ? 8 : 0);
      ctx[((size_t)(b * Sn + row)) * Dn + hoff + n * 16 + l16] =
          f2bf(co[n][e] / lrow[e]);
    }
}

// ---------------------------------------------------------------------------
// out = LayerNorm(x + y) * g + be  (TF style); optional extra bf16 copy.
// ---------------------------------------------------------------------------
template <bool DUAL>
__global__ __launch_bounds__(256)
void ln_residual_kernel(const float* __restrict__ x, const float* __restrict__ y,
                        const float* __restrict__ g, const float* __restrict__ be,
                        float* __restrict__ out, __bf16* __restrict__ outb) {
  const int row = blockIdx.x;
  const int tid = threadIdx.x;
  const float* xr = x + (size_t)row * Dn;
  const float* yr = y + (size_t)row * Dn;

  float v[4];
  float s = 0.f;
#pragma unroll
  for (int i = 0; i < 4; ++i) {
    int c = tid + i * 256;
    v[i] = xr[c] + yr[c];
    s += v[i];
  }
  __shared__ float red[256];
  red[tid] = s;
  __syncthreads();
  for (int o = 128; o > 0; o >>= 1) {
    if (tid < o) red[tid] += red[tid + o];
    __syncthreads();
  }
  const float mean = red[0] * (1.f / Dn);
  __syncthreads();

  float s2 = 0.f;
#pragma unroll
  for (int i = 0; i < 4; ++i) {
    float d = v[i] - mean;
    s2 += d * d;
  }
  red[tid] = s2;
  __syncthreads();
  for (int o = 128; o > 0; o >>= 1) {
    if (tid < o) red[tid] += red[tid + o];
    __syncthreads();
  }
  const float var  = red[0] * (1.f / Dn);
  const float rstd = rsqrtf(var + 1e-12f);

#pragma unroll
  for (int i = 0; i < 4; ++i) {
    int c = tid + i * 256;
    float r = g[c] * (v[i] - mean) * rstd + be[c];
    out[(size_t)row * Dn + c] = r;
    if (DUAL) outb[(size_t)row * Dn + c] = f2bf(r);
  }
}

// ---------------------------------------------------------------------------
extern "C" void kernel_launch(void* const* d_in, const int* in_sizes, int n_in,
                              void* d_out, int out_size, void* d_ws, size_t ws_size,
                              hipStream_t stream) {
  (void)in_sizes; (void)n_in; (void)out_size; (void)ws_size;

  const float* x    = (const float*)d_in[0];
  const int*   mask = (const int*)  d_in[1];
  const float* wq   = (const float*)d_in[2];
  const float* bq   = (const float*)d_in[3];
  const float* wk   = (const float*)d_in[4];
  const float* bk   = (const float*)d_in[5];
  const float* wv   = (const float*)d_in[6];
  const float* bv   = (const float*)d_in[7];
  const float* wo   = (const float*)d_in[8];
  const float* bo   = (const float*)d_in[9];
  const float* g1   = (const float*)d_in[10];
  const float* be1  = (const float*)d_in[11];
  const float* w1   = (const float*)d_in[12];
  const float* bf1  = (const float*)d_in[13];
  const float* w2   = (const float*)d_in[14];
  const float* bf2  = (const float*)d_in[15];
  const float* g2   = (const float*)d_in[16];
  const float* be2  = (const float*)d_in[17];
  float* out = (float*)d_out;

  // workspace layout (bytes)
  char* p = (char*)d_ws;
  auto take = [&](size_t bytes) { char* r = p; p += (bytes + 255) & ~(size_t)255; return r; };
  __bf16* xb   = (__bf16*)take((size_t)BSDn * 2);
  __bf16* wqt  = (__bf16*)take((size_t)Dn * Dn * 2);
  __bf16* wkt  = (__bf16*)take((size_t)Dn * Dn * 2);
  __bf16* wvt  = (__bf16*)take((size_t)Dn * Dn * 2);
  __bf16* wot  = (__bf16*)take((size_t)Dn * Dn * 2);
  __bf16* w1t  = (__bf16*)take((size_t)Dn * FFn * 2);
  __bf16* w2t  = (__bf16*)take((size_t)Dn * FFn * 2);
  __bf16* qbb  = (__bf16*)take((size_t)BSDn * 2);
  __bf16* kbb  = (__bf16*)take((size_t)BSDn * 2);
  __bf16* vtt  = (__bf16*)take((size_t)BSDn * 2);        // [D][token]
  __bf16* ctxb = (__bf16*)take((size_t)BSDn * 2);
  __bf16* hbb  = (__bf16*)take((size_t)BSDn * 2);
  __bf16* ffnb = (__bf16*)take((size_t)TOKn * FFn * 2);
  float*  tmp1 = (float*) take((size_t)BSDn * 4);
  float*  tmp2 = (float*) take((size_t)BSDn * 4);
  float*  hb   = (float*) take((size_t)BSDn * 4);

  const int M = TOKn;  // 4096
  dim3 tb(32, 8);

  // --- one-time layout pass: bf16 activations + transposed bf16 weights ---
  cvt_bf16_kernel<<<BSDn / 1024, 256, 0, stream>>>(x, xb);
  transpose_cvt_kernel<<<dim3(Dn / 32, Dn / 32), tb, 0, stream>>>(wq, wqt, Dn, Dn);
  transpose_cvt_kernel<<<dim3(Dn / 32, Dn / 32), tb, 0, stream>>>(wk, wkt, Dn, Dn);
  transpose_cvt_kernel<<<dim3(Dn / 32, Dn / 32), tb, 0, stream>>>(wv, wvt, Dn, Dn);
  transpose_cvt_kernel<<<dim3(Dn / 32, Dn / 32), tb, 0, stream>>>(wo, wot, Dn, Dn);
  transpose_cvt_kernel<<<dim3(FFn / 32, Dn / 32), tb, 0, stream>>>(w1, w1t, Dn, FFn);
  transpose_cvt_kernel<<<dim3(Dn / 32, FFn / 32), tb, 0, stream>>>(w2, w2t, FFn, Dn);

  dim3 gD(Dn / 64, M / 64);     // N=1024: (16, 64)
  dim3 gF(FFn / 64, M / 64);    // N=4096: (64, 64)

  // Q, K, V projections (V stored transposed for attention)
  gemm_bias_kernel<false, OUT_BF16  ><<<gD, 32, 0, stream>>>(xb, wqt, bq, qbb, M, Dn, Dn);
  gemm_bias_kernel<false, OUT_BF16  ><<<gD, 32, 0, stream>>>(xb, wkt, bk, kbb, M, Dn, Dn);
  gemm_bias_kernel<false, OUT_BF16_T><<<gD, 32, 0, stream>>>(xb, wvt, bv, vtt, M, Dn, Dn);

  // Flash attention
  attn_kernel<<<dim3(Sn / 16, Hn, Bn), 32, 0, stream>>>(qbb, kbb, vtt, mask, ctxb);

  // Output projection, residual + LN1 (f32 + bf16 copy for FFN)
  gemm_bias_kernel<false, OUT_F32><<<gD, 32, 0, stream>>>(ctxb, wot, bo, tmp1, M, Dn, Dn);
  ln_residual_kernel<true><<<M, 256, 0, stream>>>(x, tmp1, g1, be1, hb, hbb);

  // FFN
  gemm_bias_kernel<true,  OUT_BF16><<<gF, 32, 0, stream>>>(hbb, w1t, bf1, ffnb, M, FFn, Dn);
  gemm_bias_kernel<false, OUT_F32 ><<<gD, 32, 0, stream>>>(ffnb, w2t, bf2, tmp2, M, Dn, FFn);
  ln_residual_kernel<false><<<M, 256, 0, stream>>>(hb, tmp2, g2, be2, out, nullptr);
}
